// RVQModel_32358283608539
// MI455X (gfx1250) — compile-verified, matrix operand order checked by reading, not run on hardware
//
#include <hip/hip_runtime.h>
#include <hip/hip_bf16.h>
#include <math.h>

// ---------------------------------------------------------------------------
// Types for CDNA5 WMMA
// ---------------------------------------------------------------------------
typedef __attribute__((ext_vector_type(16))) __bf16 v16bf;
typedef __attribute__((ext_vector_type(8)))  __bf16 v8bf;
typedef __attribute__((ext_vector_type(8)))  float  v8f;

// Model dims (fixed by the reference)
static constexpr int Bb = 32, Tt = 512, Cc = 256, Dd = 1024, Hh = 16, Ll = 6;
static constexpr int FFd = 4096, Oo = 256, Qq = 8, Kk = 1024, HD = 64;
static constexpr int Mrows = Bb * Tt; // 16384

// ---------------------------------------------------------------------------
// Fragment loader for V_WMMA_F32_16X16X32_BF16 (A-operand layout; B uses the
// same layout when the tile is staged transposed as [N][K] in LDS).
// Per lane (16-bit A 16x32, ISA 7.12.2):
//   lanes 0-15 : row = lane,    K = [0..7] then [16..23]
//   lanes 16-31: row = lane-16, K = [8..15] then [24..31]
// ---------------------------------------------------------------------------
__device__ inline v16bf load_frag(const __bf16* base, int rowBase, int ld,
                                  int koff, int lane) {
  const int half8 = (lane >> 4) << 3;            // 0 or 8
  const __bf16* p = base + (size_t)(rowBase + (lane & 15)) * ld + koff + half8;
  v8bf r1 = *(const v8bf*)p;                      // K = koff+half8 .. +7
  v8bf r2 = *(const v8bf*)(p + 16);               // K = koff+half8+16 .. +7
  v16bf out;
  ((v8bf*)&out)[0] = r1;
  ((v8bf*)&out)[1] = r2;
  return out;
}

__device__ inline float gelu_exact(float x) {
  return 0.5f * x * (1.0f + erff(x * 0.70710678118654752f));
}

// ---------------------------------------------------------------------------
// Generic GEMM:  out[M,N] = epilogue( A[M,K] @ Bw[K,N] + bias[N] (+ res) )
// A,Bw f32 in global, converted to bf16 while staging into LDS; WMMA bf16 /
// f32 accumulate.  res (optional): out += res[(row % resMod)*N + col]
// (resMod=T -> pos broadcast add, resMod=2^30 -> plain residual add).
// Block: 256 threads = 8 waves. BM=128, BN=128, BK=32 (one WMMA K-step).
// Wave (wm 0..3, wn 0..1) owns a 32x64 tile -> acc[2][4], 8 WMMAs/K-step.
// K-loop is software-pipelined: next tile's global loads are issued before
// the current tile's WMMAs.
// ---------------------------------------------------------------------------
#define GBM 128
#define GBN 128
#define GBK 32
#define GLD 40

template <bool GELU>
__global__ __launch_bounds__(256) void gemm_bf16_kernel(
    const float* __restrict__ A, const float* __restrict__ Bw,
    const float* __restrict__ bias, const float* __restrict__ res, int resMod,
    float* __restrict__ out, int M, int N, int K) {
  __shared__ __align__(16) __bf16 As[GBM][GLD];
  __shared__ __align__(16) __bf16 Bs[GBN][GLD];   // transposed: Bs[n][k]

  const int t = threadIdx.x;
  const int lane = t & 31;
  const int wave = t >> 5;          // 0..7
  const int wm = wave >> 1;         // 0..3  (rows wm*32)
  const int wn = wave & 1;          // 0..1  (cols wn*64)
  const int m0 = blockIdx.y * GBM;
  const int n0 = blockIdx.x * GBN;

  v8f acc[2][4] = {};
  float4 ar[4], br[4];

  // A tile: 128x32 f32 -> 1024 float4 chunks; r = c/8, col = (c%8)*4
  // B tile: 32x128 f32 -> 1024 float4 chunks; kk = c/32, nn = (c%32)*4
  auto loadA = [&](int k0) {
#pragma unroll
    for (int it = 0; it < 4; ++it) {
      int c = it * 256 + t;
      int r = c >> 3;
      int col = (c & 7) << 2;
      int gr = m0 + r;
      ar[it] = make_float4(0.f, 0.f, 0.f, 0.f);
      if (gr < M) ar[it] = *(const float4*)(A + (size_t)gr * K + k0 + col);
    }
  };
  auto loadB = [&](int k0) {
#pragma unroll
    for (int it = 0; it < 4; ++it) {
      int c = it * 256 + t;
      int kk = c >> 5;
      int nn = (c & 31) << 2;
      br[it] = *(const float4*)(Bw + (size_t)(k0 + kk) * N + n0 + nn);
    }
  };
  auto stageTiles = [&]() {
#pragma unroll
    for (int it = 0; it < 4; ++it) {
      int c = it * 256 + t;
      int r = c >> 3;
      int col = (c & 7) << 2;
      As[r][col + 0] = (__bf16)ar[it].x;
      As[r][col + 1] = (__bf16)ar[it].y;
      As[r][col + 2] = (__bf16)ar[it].z;
      As[r][col + 3] = (__bf16)ar[it].w;
    }
#pragma unroll
    for (int it = 0; it < 4; ++it) {
      int c = it * 256 + t;
      int kk = c >> 5;
      int nn = (c & 31) << 2;
      Bs[nn + 0][kk] = (__bf16)br[it].x;
      Bs[nn + 1][kk] = (__bf16)br[it].y;
      Bs[nn + 2][kk] = (__bf16)br[it].z;
      Bs[nn + 3][kk] = (__bf16)br[it].w;
    }
  };

  loadA(0);
  loadB(0);
  for (int k0 = 0; k0 < K; k0 += GBK) {
    stageTiles();
    __syncthreads();
    if (k0 + GBK < K) {   // issue next tile's global loads before the math
      loadA(k0 + GBK);
      loadB(k0 + GBK);
    }
    v16bf af[2], bfv[4];
#pragma unroll
    for (int mt = 0; mt < 2; ++mt)
      af[mt] = load_frag(&As[0][0], wm * 32 + mt * 16, GLD, 0, lane);
#pragma unroll
    for (int nt = 0; nt < 4; ++nt)
      bfv[nt] = load_frag(&Bs[0][0], wn * 64 + nt * 16, GLD, 0, lane);
#pragma unroll
    for (int mt = 0; mt < 2; ++mt)
#pragma unroll
      for (int nt = 0; nt < 4; ++nt)
        acc[mt][nt] = __builtin_amdgcn_wmma_f32_16x16x32_bf16(
            false, af[mt], false, bfv[nt], (short)0, acc[mt][nt], false, false);
    __syncthreads();
  }

  // --- epilogue ---
  const int half8 = (lane >> 4) << 3;
#pragma unroll
  for (int mt = 0; mt < 2; ++mt)
#pragma unroll
    for (int nt = 0; nt < 4; ++nt) {
      int gcol = n0 + wn * 64 + nt * 16 + (lane & 15);
      int rbase = m0 + wm * 32 + mt * 16 + half8;
      float bv = bias ? bias[gcol] : 0.f;
#pragma unroll
      for (int r = 0; r < 8; ++r) {
        int gm = rbase + r;
        if (gm < M) {
          float v = acc[mt][nt][r] + bv;
          if (res) v += res[(size_t)(gm % resMod) * N + gcol];
          if (GELU) v = gelu_exact(v);
          out[(size_t)gm * N + gcol] = v;
        }
      }
    }
}

// ---------------------------------------------------------------------------
// LayerNorm, stride-parameterized (lets us LN only the 32 last-token rows).
// One block (256 thr) per row of width D=1024.
// ---------------------------------------------------------------------------
__global__ __launch_bounds__(256) void ln_kernel(
    const float* __restrict__ in, float* __restrict__ out,
    const float* __restrict__ g, const float* __restrict__ b, int D,
    size_t strideIn, size_t strideOut) {
  const float* x = in + (size_t)blockIdx.x * strideIn;
  float* y = out + (size_t)blockIdx.x * strideOut;
  const int t = threadIdx.x;
  float s = 0.f, s2 = 0.f;
  for (int i = t; i < D; i += 256) {
    float v = x[i];
    s += v;
    s2 += v * v;
  }
#pragma unroll
  for (int o = 16; o > 0; o >>= 1) {
    s += __shfl_xor(s, o, 32);
    s2 += __shfl_xor(s2, o, 32);
  }
  __shared__ float w1[8], w2[8];
  if ((t & 31) == 0) {
    w1[t >> 5] = s;
    w2[t >> 5] = s2;
  }
  __syncthreads();
  float S = 0.f, S2 = 0.f;
#pragma unroll
  for (int i = 0; i < 8; ++i) {
    S += w1[i];
    S2 += w2[i];
  }
  float mean = S / D;
  float var = S2 / D - mean * mean;
  float rstd = rsqrtf(var + 1e-5f);
  for (int i = t; i < D; i += 256) y[i] = (x[i] - mean) * rstd * g[i] + b[i];
}

// ---------------------------------------------------------------------------
// Flash attention, causal, per (b, h) with 64-query blocks.
// Block = 128 threads = 4 waves; wave w owns 16 query rows (qw = q0+16w).
// Per 32-key chunk: S(16x32) via 2x2 WMMA, online softmax with shfl-xor row
// reductions, P staged to wave-private LDS (s_wait_dscnt orders the DS RAW),
// O += P@V via 4 WMMAs over the 64 head dims.
// qkv layout: ((b*T+t)*3 + c)*D + h*64 + d.  Output -> a[(b*T+t)*D + h*64+d].
// ---------------------------------------------------------------------------
__global__ __launch_bounds__(128) void attn_kernel(
    const float* __restrict__ qkv, float* __restrict__ out) {
  __shared__ __align__(16) __bf16 Qs[64][80];
  __shared__ __align__(16) __bf16 Ks[32][80];
  __shared__ __align__(16) __bf16 Vt[64][40];      // Vt[d][key]
  __shared__ __align__(16) __bf16 Ps[4][16][40];   // per-wave P (16x32)

  const int t = threadIdx.x;
  const int lane = t & 31;
  const int wave = t >> 5;
  const int bh = blockIdx.y;
  const int b = bh >> 4;     // H = 16
  const int hh = bh & 15;
  const int q0 = blockIdx.x * 64;
  const float scale = 0.125f;  // 1/sqrt(64)

  // load + scale Q tile (64x64)
#pragma unroll
  for (int it = 0; it < 8; ++it) {
    int c = it * 128 + t;
    int r = c >> 4;
    int d4 = (c & 15) << 2;
    float4 v = *(const float4*)(qkv + ((size_t)(b * Tt + q0 + r) * 3 + 0) * Dd +
                                hh * HD + d4);
    Qs[r][d4 + 0] = (__bf16)(v.x * scale);
    Qs[r][d4 + 1] = (__bf16)(v.y * scale);
    Qs[r][d4 + 2] = (__bf16)(v.z * scale);
    Qs[r][d4 + 3] = (__bf16)(v.w * scale);
  }

  v8f o[4] = {};
  float mrow[8], lrow[8];
#pragma unroll
  for (int r = 0; r < 8; ++r) {
    mrow[r] = -1e30f;
    lrow[r] = 0.f;
  }
  const int qw = q0 + wave * 16;
  const int half8 = (lane >> 4) << 3;
  const int colb = lane & 15;
  const int nch = q0 / 32 + 2;  // chunks covering keys <= q0+63

  for (int kc = 0; kc < nch; ++kc) {
    const int k0 = kc * 32;
    // cooperative stage of K (32x64) and V^T (64x32)
#pragma unroll
    for (int it = 0; it < 4; ++it) {
      int c = it * 128 + t;
      int r = c >> 4;
      int d4 = (c & 15) << 2;
      float4 kv = *(const float4*)(qkv +
                                   ((size_t)(b * Tt + k0 + r) * 3 + 1) * Dd +
                                   hh * HD + d4);
      Ks[r][d4 + 0] = (__bf16)kv.x;
      Ks[r][d4 + 1] = (__bf16)kv.y;
      Ks[r][d4 + 2] = (__bf16)kv.z;
      Ks[r][d4 + 3] = (__bf16)kv.w;
      float4 vv = *(const float4*)(qkv +
                                   ((size_t)(b * Tt + k0 + r) * 3 + 2) * Dd +
                                   hh * HD + d4);
      Vt[d4 + 0][r] = (__bf16)vv.x;
      Vt[d4 + 1][r] = (__bf16)vv.y;
      Vt[d4 + 2][r] = (__bf16)vv.z;
      Vt[d4 + 3][r] = (__bf16)vv.w;
    }
    __syncthreads();

    if (k0 <= qw + 15) {  // chunk has at least one unmasked key for this wave
      v8f s[2] = {};
#pragma unroll
      for (int ks = 0; ks < 2; ++ks) {
        v16bf aq = load_frag(&Qs[0][0], wave * 16, 80, ks * 32, lane);
#pragma unroll
        for (int nt = 0; nt < 2; ++nt) {
          v16bf bk = load_frag(&Ks[0][0], nt * 16, 80, ks * 32, lane);
          s[nt] = __builtin_amdgcn_wmma_f32_16x16x32_bf16(
              false, aq, false, bk, (short)0, s[nt], false, false);
        }
      }
#pragma unroll
      for (int r = 0; r < 8; ++r) {
        const int qrow = qw + r + half8;
        float s0 = s[0][r];
        float s1 = s[1][r];
        if (k0 + colb > qrow) s0 = -1e30f;        // causal mask
        if (k0 + 16 + colb > qrow) s1 = -1e30f;
        float rm = fmaxf(s0, s1);
        rm = fmaxf(rm, __shfl_xor(rm, 1, 32));
        rm = fmaxf(rm, __shfl_xor(rm, 2, 32));
        rm = fmaxf(rm, __shfl_xor(rm, 4, 32));
        rm = fmaxf(rm, __shfl_xor(rm, 8, 32));
        float mnew = fmaxf(mrow[r], rm);
        float alpha = expf(mrow[r] - mnew);
        float p0 = expf(s0 - mnew);
        float p1 = expf(s1 - mnew);
        float rs = p0 + p1;
        rs += __shfl_xor(rs, 1, 32);
        rs += __shfl_xor(rs, 2, 32);
        rs += __shfl_xor(rs, 4, 32);
        rs += __shfl_xor(rs, 8, 32);
        lrow[r] = lrow[r] * alpha + rs;
        mrow[r] = mnew;
        Ps[wave][r + half8][colb] = (__bf16)p0;
        Ps[wave][r + half8][16 + colb] = (__bf16)p1;
#pragma unroll
        for (int j = 0; j < 4; ++j) o[j][r] *= alpha;
      }
      // order the same-wave LDS write->read of Ps (cross-lane RAW)
      asm volatile("s_wait_dscnt 0" ::: "memory");
      v16bf ap = load_frag(&Ps[wave][0][0], 0, 40, 0, lane);
#pragma unroll
      for (int j = 0; j < 4; ++j) {
        v16bf bv = load_frag(&Vt[0][0], j * 16, 40, 0, lane);
        o[j] = __builtin_amdgcn_wmma_f32_16x16x32_bf16(
            false, ap, false, bv, (short)0, o[j], false, false);
      }
    }
    __syncthreads();
  }

#pragma unroll
  for (int j = 0; j < 4; ++j) {
    int gcol = hh * HD + j * 16 + colb;
#pragma unroll
    for (int r = 0; r < 8; ++r) {
      int qrow = qw + r + half8;
      out[(size_t)(b * Tt + qrow) * Dd + gcol] = o[j][r] / lrow[r];
    }
  }
}

// ---------------------------------------------------------------------------
// RVQ head: per batch row (32 blocks), all 8 stages sequentially.
// dist = ||r - e||^2; argmin with first-min tie-break; z_q_sum +=
// sigmoid(ls[q]) * e[idx]; residual -= e[idx]  (RESIDUAL_SCALE == 1).
// ---------------------------------------------------------------------------
__global__ __launch_bounds__(256) void rvq_kernel(
    const float* __restrict__ z, const float* __restrict__ cb,
    const float* __restrict__ ls, float* __restrict__ out) {
  const int n = blockIdx.x;
  const int t = threadIdx.x;
  __shared__ float res[Oo], zq[Oo];
  __shared__ float sdist[256];
  __shared__ int sidx[256];

  res[t] = z[(size_t)n * Oo + t];
  zq[t] = 0.f;
  __syncthreads();

  for (int q = 0; q < Qq; ++q) {
    const float* eb = cb + (size_t)q * Kk * Oo;
    float best = 3.4e38f;
    int bi = 0;
    for (int j = t; j < Kk; j += 256) {
      const float* e = eb + (size_t)j * Oo;
      float d = 0.f;
      for (int dd = 0; dd < Oo; ++dd) {
        float df = res[dd] - e[dd];
        d = fmaf(df, df, d);
      }
      if (d < best) {
        best = d;
        bi = j;
      }
    }
    sdist[t] = best;
    sidx[t] = bi;
    __syncthreads();
    for (int off = 128; off > 0; off >>= 1) {
      if (t < off) {
        if (sdist[t + off] < sdist[t] ||
            (sdist[t + off] == sdist[t] && sidx[t + off] < sidx[t])) {
          sdist[t] = sdist[t + off];
          sidx[t] = sidx[t + off];
        }
      }
      __syncthreads();
    }
    const int idx = sidx[0];
    const float sg = 1.f / (1.f + expf(-ls[q]));
    const float ev = eb[(size_t)idx * Oo + t];
    __syncthreads();  // everyone read sidx[0] before it is clobbered
    zq[t] += sg * ev;
    res[t] -= ev;
    __syncthreads();
  }
  out[(size_t)n * Oo + t] = zq[t];
}

// ---------------------------------------------------------------------------
// Host orchestration
// ---------------------------------------------------------------------------
extern "C" void kernel_launch(void* const* d_in, const int* in_sizes, int n_in,
                              void* d_out, int out_size, void* d_ws,
                              size_t ws_size, hipStream_t stream) {
  (void)in_sizes; (void)n_in; (void)out_size; (void)ws_size;

  const float* x       = (const float*)d_in[0];
  const float* w_in    = (const float*)d_in[1];
  const float* b_in    = (const float*)d_in[2];
  const float* pos     = (const float*)d_in[3];
  const float* ln1_g   = (const float*)d_in[4];
  const float* ln1_b   = (const float*)d_in[5];
  const float* qkv_w   = (const float*)d_in[6];
  const float* qkv_b   = (const float*)d_in[7];
  const float* proj_w  = (const float*)d_in[8];
  const float* proj_b  = (const float*)d_in[9];
  const float* ln2_g   = (const float*)d_in[10];
  const float* ln2_b   = (const float*)d_in[11];
  const float* ff1_w   = (const float*)d_in[12];
  const float* ff1_b   = (const float*)d_in[13];
  const float* ff2_w   = (const float*)d_in[14];
  const float* ff2_b   = (const float*)d_in[15];
  const float* lnf_g   = (const float*)d_in[16];
  const float* lnf_b   = (const float*)d_in[17];
  const float* out1_w  = (const float*)d_in[18];
  const float* out1_b  = (const float*)d_in[19];
  const float* out2_w  = (const float*)d_in[20];
  const float* out2_b  = (const float*)d_in[21];
  const float* codebooks = (const float*)d_in[22];
  const float* lscales   = (const float*)d_in[23];

  float* ws = (float*)d_ws;
  size_t off = 0;
  float* h   = ws + off; off += (size_t)Mrows * Dd;    // residual stream
  float* a   = ws + off; off += (size_t)Mrows * Dd;    // LN out / attn out
  float* qkv = ws + off; off += (size_t)Mrows * 3 * Dd;
  float* ffm = ws + off; off += (size_t)Mrows * FFd;
  float* z0  = ws + off; off += (size_t)Bb * Dd;
  float* g1  = ws + off; off += (size_t)Bb * Dd;
  float* zz  = ws + off; off += (size_t)Bb * Oo;

  const int BIGMOD = 1 << 30;
  auto ggrid = [](int M, int N) { return dim3((N + GBN - 1) / GBN,
                                              (M + GBM - 1) / GBM); };
  dim3 blk(256);

  // h = x @ w_in + b_in + pos[t]
  gemm_bf16_kernel<false><<<ggrid(Mrows, Dd), blk, 0, stream>>>(
      x, w_in, b_in, pos, Tt, h, Mrows, Dd, Cc);

  for (int l = 0; l < Ll; ++l) {
    // a = LN1(h)
    ln_kernel<<<Mrows, blk, 0, stream>>>(h, a, ln1_g + (size_t)l * Dd,
                                         ln1_b + (size_t)l * Dd, Dd, Dd, Dd);
    // qkv = a @ qkv_w + qkv_b
    gemm_bf16_kernel<false><<<ggrid(Mrows, 3 * Dd), blk, 0, stream>>>(
        a, qkv_w + (size_t)l * Dd * 3 * Dd, qkv_b + (size_t)l * 3 * Dd,
        nullptr, 1, qkv, Mrows, 3 * Dd, Dd);
    // a = attention(qkv)
    attn_kernel<<<dim3(Tt / 64, Bb * Hh), dim3(128), 0, stream>>>(qkv, a);
    // h = h + a @ proj_w + proj_b
    gemm_bf16_kernel<false><<<ggrid(Mrows, Dd), blk, 0, stream>>>(
        a, proj_w + (size_t)l * Dd * Dd, proj_b + (size_t)l * Dd, h, BIGMOD,
        h, Mrows, Dd, Dd);
    // a = LN2(h)
    ln_kernel<<<Mrows, blk, 0, stream>>>(h, a, ln2_g + (size_t)l * Dd,
                                         ln2_b + (size_t)l * Dd, Dd, Dd, Dd);
    // ffm = gelu(a @ ff1_w + ff1_b)
    gemm_bf16_kernel<true><<<ggrid(Mrows, FFd), blk, 0, stream>>>(
        a, ff1_w + (size_t)l * Dd * FFd, ff1_b + (size_t)l * FFd, nullptr, 1,
        ffm, Mrows, FFd, Dd);
    // h = h + ffm @ ff2_w + ff2_b
    gemm_bf16_kernel<false><<<ggrid(Mrows, Dd), blk, 0, stream>>>(
        ffm, ff2_w + (size_t)l * FFd * Dd, ff2_b + (size_t)l * Dd, h, BIGMOD,
        h, Mrows, Dd, FFd);
  }

  // z0 = LNf(h[:, -1, :])  (32 rows, stride T*D)
  ln_kernel<<<Bb, blk, 0, stream>>>(h + (size_t)(Tt - 1) * Dd, z0, lnf_g,
                                    lnf_b, Dd, (size_t)Tt * Dd, Dd);
  // g1 = gelu(z0 @ out1_w + out1_b)
  gemm_bf16_kernel<true><<<ggrid(Bb, Dd), blk, 0, stream>>>(
      z0, out1_w, out1_b, nullptr, 1, g1, Bb, Dd, Dd);
  // zz = g1 @ out2_w + out2_b
  gemm_bf16_kernel<false><<<ggrid(Bb, Oo), blk, 0, stream>>>(
      g1, out2_w, out2_b, nullptr, 1, zz, Bb, Oo, Dd);
  // RVQ -> d_out
  rvq_kernel<<<Bb, blk, 0, stream>>>(zz, codebooks, lscales, (float*)d_out);
}
